// BertTempRel_67190468378649
// MI455X (gfx1250) — matrix operand order repaired and checked by reading, not compiled, not run on hardware
//
#include <hip/hip_runtime.h>
#include <hip/hip_bf16.h>

typedef _Float16 v16h __attribute__((ext_vector_type(16)));
typedef _Float16 v8h  __attribute__((ext_vector_type(8)));
typedef _Float16 v4h  __attribute__((ext_vector_type(4)));
typedef float    v8f  __attribute__((ext_vector_type(8)));

#define BB 1024
#define SS 512
#define DD 768
#define F1 2304   // 3*D
#define H1 256
#define H2 64
#define NC 4

// ---------------------------------------------------------------------------
// Kernel 1: span means + cls gather -> feat (f16).  HBM-bound part.
// One block per batch row; 192 threads, each owns one float4 (4 dims) of D=768.
// ---------------------------------------------------------------------------
__global__ __launch_bounds__(192) void span_mean_kernel(
    const float* __restrict__ x, const int* __restrict__ e1s,
    const int* __restrict__ e2s, _Float16* __restrict__ feat) {
  const int b = blockIdx.x;
  const int t = threadIdx.x;              // 0..191, float4 slice index
  const float4* xb = (const float4*)(x + (size_t)b * SS * DD);

  int lo1 = e1s[2 * b];
  int hi1 = e1s[2 * b + 1]; hi1 = hi1 > lo1 + 1 ? hi1 : lo1 + 1;
  int lo2 = e2s[2 * b];
  int hi2 = e2s[2 * b + 1]; hi2 = hi2 > lo2 + 1 ? hi2 : lo2 + 1;

  float a1x = 0.f, a1y = 0.f, a1z = 0.f, a1w = 0.f;
  for (int s = lo1; s < hi1; ++s) {
    float4 v = xb[(size_t)s * (DD / 4) + t];
    a1x += v.x; a1y += v.y; a1z += v.z; a1w += v.w;
  }
  float a2x = 0.f, a2y = 0.f, a2z = 0.f, a2w = 0.f;
  for (int s = lo2; s < hi2; ++s) {
    float4 v = xb[(size_t)s * (DD / 4) + t];
    a2x += v.x; a2y += v.y; a2z += v.z; a2w += v.w;
  }
  const float i1 = 1.f / (float)(hi1 - lo1);
  const float i2 = 1.f / (float)(hi2 - lo2);
  float4 cls = xb[t];

  _Float16* fb = feat + (size_t)b * F1;
  v4h o;
  o[0] = (_Float16)(a1x * i1); o[1] = (_Float16)(a1y * i1);
  o[2] = (_Float16)(a1z * i1); o[3] = (_Float16)(a1w * i1);
  *(v4h*)(fb + 4 * t) = o;
  o[0] = (_Float16)(a2x * i2); o[1] = (_Float16)(a2y * i2);
  o[2] = (_Float16)(a2z * i2); o[3] = (_Float16)(a2w * i2);
  *(v4h*)(fb + DD + 4 * t) = o;
  o[0] = (_Float16)cls.x; o[1] = (_Float16)cls.y;
  o[2] = (_Float16)cls.z; o[3] = (_Float16)cls.w;
  *(v4h*)(fb + 2 * DD + 4 * t) = o;
}

// ---------------------------------------------------------------------------
// Kernel 2: W1 (2304x256) -> W1t f16 (256x2304), W2 (256x64) -> W2t f16 (64x256)
// Transposed so WMMA B-fragment loads are contiguous along K.
// ---------------------------------------------------------------------------
__global__ void convert_weights(const float* __restrict__ W1,
                                const float* __restrict__ W2,
                                _Float16* __restrict__ W1t,
                                _Float16* __restrict__ W2t) {
  const int total1 = H1 * F1;        // 589824
  const int total2 = H2 * H1;        // 16384
  for (int i = blockIdx.x * blockDim.x + threadIdx.x; i < total1 + total2;
       i += gridDim.x * blockDim.x) {
    if (i < total1) {
      int n = i / F1, k = i % F1;
      W1t[i] = (_Float16)W1[(size_t)k * H1 + n];
    } else {
      int j = i - total1;
      int n = j / H1, k = j % H1;
      W2t[j] = (_Float16)W2[(size_t)k * H2 + n];
    }
  }
}

// ---------------------------------------------------------------------------
// WMMA fragment loaders (contiguous-K order chosen by this code, used for both A and B, wave32).
// A 16x32 f16: lane -> M = lane%16 ; elements e: K = 16*(lane/16) + e.
// B 32x16 f16: lane -> N = lane%16 ; elements e: K = 16*(lane/16) + e.
// ---------------------------------------------------------------------------
__device__ __forceinline__ v16h load_a_frag(const _Float16* __restrict__ A,
                                            int ldk, int row, int khalf, int kk) {
  const _Float16* p = A + (size_t)row * ldk + kk + khalf * 16;
  v8h a0 = *(const v8h*)(p);
  v8h a1 = *(const v8h*)(p + 8);
  return __builtin_shufflevector(a0, a1, 0, 1, 2, 3, 4, 5, 6, 7,
                                 8, 9, 10, 11, 12, 13, 14, 15);
}

__device__ __forceinline__ v16h load_b_frag(const _Float16* __restrict__ Bt,
                                            int ldk, int col, int khalf, int kk) {
  return *(const v16h*)(Bt + (size_t)col * ldk + kk + khalf * 16);
}

// ---------------------------------------------------------------------------
// Kernel 3: h1 = relu(feat @ W1 + b1)  (1024x2304)@(2304x256)
// 4 waves / block; each wave one 16x16 tile. grid = (M/16, N/64).
// ---------------------------------------------------------------------------
__global__ __launch_bounds__(128) void gemm1_relu(
    const _Float16* __restrict__ A, const _Float16* __restrict__ Bt,
    const float* __restrict__ bias, _Float16* __restrict__ C) {
  const int lane = threadIdx.x & 31;
  const int wave = threadIdx.x >> 5;
  const int mBase = blockIdx.x * 16;
  const int nBase = (blockIdx.y * 4 + wave) * 16;
  const int row = mBase + (lane & 15);
  const int col = nBase + (lane & 15);
  const int khalf = lane >> 4;

  v8f c = {};
  for (int kk = 0; kk < F1; kk += 32) {
    v16h a = load_a_frag(A, F1, row, khalf, kk);
    v16h b = load_b_frag(Bt, F1, col, khalf, kk);
    c = __builtin_amdgcn_wmma_f32_16x16x32_f16(false, a, false, b,
                                               (short)0, c, false, false);
  }
  const float bn = bias[col];
  const int mOff = khalf * 8;
#pragma unroll
  for (int r = 0; r < 8; ++r) {
    float v = c[r] + bn;
    v = v > 0.f ? v : 0.f;
    C[(size_t)(mBase + mOff + r) * H1 + col] = (_Float16)v;
  }
}

// ---------------------------------------------------------------------------
// Kernel 4: h2 = relu(h1 @ W2 + b2)  (1024x256)@(256x64) -> f32
// ---------------------------------------------------------------------------
__global__ __launch_bounds__(128) void gemm2_relu(
    const _Float16* __restrict__ A, const _Float16* __restrict__ Bt,
    const float* __restrict__ bias, float* __restrict__ C) {
  const int lane = threadIdx.x & 31;
  const int wave = threadIdx.x >> 5;     // 4 n-tiles cover N=64
  const int mBase = blockIdx.x * 16;
  const int nBase = wave * 16;
  const int row = mBase + (lane & 15);
  const int col = nBase + (lane & 15);
  const int khalf = lane >> 4;

  v8f c = {};
#pragma unroll
  for (int kk = 0; kk < H1; kk += 32) {
    v16h a = load_a_frag(A, H1, row, khalf, kk);
    v16h b = load_b_frag(Bt, H1, col, khalf, kk);
    c = __builtin_amdgcn_wmma_f32_16x16x32_f16(false, a, false, b,
                                               (short)0, c, false, false);
  }
  const float bn = bias[col];
  const int mOff = khalf * 8;
#pragma unroll
  for (int r = 0; r < 8; ++r) {
    float v = c[r] + bn;
    C[(size_t)(mBase + mOff + r) * H2 + col] = v > 0.f ? v : 0.f;
  }
}

// ---------------------------------------------------------------------------
// Kernel 5: logits = h2 @ W3 + b3, softmax over 4. One thread per batch row.
// ---------------------------------------------------------------------------
__global__ void head_softmax(const float* __restrict__ H,
                             const float* __restrict__ W3,
                             const float* __restrict__ b3,
                             float* __restrict__ out) {
  const int b = blockIdx.x * blockDim.x + threadIdx.x;
  if (b >= BB) return;
  const float* h = H + (size_t)b * H2;
  float acc0 = b3[0], acc1 = b3[1], acc2 = b3[2], acc3 = b3[3];
#pragma unroll 8
  for (int k = 0; k < H2; ++k) {
    const float hv = h[k];
    acc0 += hv * W3[k * NC + 0];
    acc1 += hv * W3[k * NC + 1];
    acc2 += hv * W3[k * NC + 2];
    acc3 += hv * W3[k * NC + 3];
  }
  float m = fmaxf(fmaxf(acc0, acc1), fmaxf(acc2, acc3));
  float e0 = __expf(acc0 - m), e1 = __expf(acc1 - m);
  float e2 = __expf(acc2 - m), e3 = __expf(acc3 - m);
  float inv = 1.f / (e0 + e1 + e2 + e3);
  out[b * NC + 0] = e0 * inv;
  out[b * NC + 1] = e1 * inv;
  out[b * NC + 2] = e2 * inv;
  out[b * NC + 3] = e3 * inv;
}

// ---------------------------------------------------------------------------
extern "C" void kernel_launch(void* const* d_in, const int* in_sizes, int n_in,
                              void* d_out, int out_size, void* d_ws, size_t ws_size,
                              hipStream_t stream) {
  const float* x   = (const float*)d_in[0];
  const int* e1s   = (const int*)d_in[1];
  const int* e2s   = (const int*)d_in[2];
  const float* W1  = (const float*)d_in[3];
  const float* b1  = (const float*)d_in[4];
  const float* W2  = (const float*)d_in[5];
  const float* b2  = (const float*)d_in[6];
  const float* W3  = (const float*)d_in[7];
  const float* b3  = (const float*)d_in[8];
  float* out = (float*)d_out;

  // Workspace layout (all offsets 256B aligned)
  char* ws = (char*)d_ws;
  _Float16* featH = (_Float16*)(ws);                     // 1024*2304*2 = 4718592
  _Float16* W1t   = (_Float16*)(ws + 4718592);           // 256*2304*2  = 1179648
  _Float16* W2t   = (_Float16*)(ws + 4718592 + 1179648); // 64*256*2    = 32768
  _Float16* h1H   = (_Float16*)(ws + 4718592 + 1179648 + 32768);   // 1024*256*2 = 524288
  float*    h2F   = (float*)   (ws + 4718592 + 1179648 + 32768 + 524288); // 1024*64*4

  // 1. span means + cls -> f16 feature matrix (bandwidth-dominant)
  span_mean_kernel<<<BB, 192, 0, stream>>>(x, e1s, e2s, featH);

  // 2. weight cvt+transpose (runs concurrently-ordered on same stream)
  convert_weights<<<512, 256, 0, stream>>>(W1, W2, W1t, W2t);

  // 3. feat @ W1 + b1, relu   (64 x 4 blocks, 4 waves each -> 16x16 tiles)
  gemm1_relu<<<dim3(BB / 16, H1 / 64), 128, 0, stream>>>(featH, W1t, b1, h1H);

  // 4. h1 @ W2 + b2, relu
  gemm2_relu<<<dim3(BB / 16, 1), 128, 0, stream>>>(h1H, W2t, b2, h2F);

  // 5. h2 @ W3 + b3, softmax
  head_softmax<<<(BB + 255) / 256, 256, 0, stream>>>(h2F, W3, b3, out);
}
